// SoftmaxAttention_58626303590885
// MI455X (gfx1250) — matrix operand order repaired and checked by reading, not compiled, not run on hardware
//
#include <hip/hip_runtime.h>

typedef __attribute__((ext_vector_type(16))) _Float16 v16h;
typedef __attribute__((ext_vector_type(8)))  _Float16 v8h;
typedef __attribute__((ext_vector_type(8)))  float    v8f;
typedef __attribute__((ext_vector_type(4)))  unsigned int u32x4;
typedef __attribute__((ext_vector_type(4)))  int      i32x4;
typedef __attribute__((ext_vector_type(8)))  int      i32x8;

// Problem dims (from reference): B=32, P=512, H=512, D=1024
static constexpr int BB = 32;
static constexpr int PP = 512;
static constexpr int HH = 512;
static constexpr int DD = 1024;

// ---------------------------------------------------------------------------
// Issue one TDM 2D tile load: 128 rows x 32 f16 from a row-major [rows, K]
// operand into LDS (packed 128x32, row stride 64B).  D# per ISA ch.8:
//   group0: count=1 | lds_addr | global_addr | type=2
//   group1: data_size=2B, tensor_dim0=K, tensor_dim1=128, tile_dim0=32,
//           tile_dim1=128, tensor_dim0_stride=K
// Groups 2/3 zero (2D tile: all-zero encodes "unused").
// ---------------------------------------------------------------------------
__device__ __forceinline__ void tdm_load_tile_2d(unsigned lds_addr,
                                                 const _Float16* gptr, int K) {
  const unsigned long long ga = (unsigned long long)(uintptr_t)gptr;
  u32x4 g0;
  g0[0] = 1u;                                   // count=1, user descriptor
  g0[1] = lds_addr;                             // LDS byte address
  g0[2] = (unsigned)ga;                         // global addr [31:0]
  g0[3] = (unsigned)(ga >> 32) | 0x80000000u;   // addr [56:32] | type=2<<30
  i32x8 g1;
  g1[0] = 0x00010000;          // workgroup_mask=0, data_size=1 (2 bytes)
  g1[1] = (K & 0xFFFF) << 16;  // tensor_dim0 = K          (bits 79:48)
  g1[2] = (128 << 16);         // tensor_dim1 = 128        (bits 111:80)
  g1[3] = (32 << 16);          // tile_dim0 = 32           (bits 127:112)
  g1[4] = 128;                 // tile_dim1 = 128, tile_dim2 = 0
  g1[5] = K;                   // tensor_dim0_stride = K   (48-bit)
  g1[6] = 0;
  g1[7] = 0;
  i32x4 z4 = {0, 0, 0, 0};
#if defined(__clang_major__) && __clang_major__ >= 23
  i32x8 z8 = {0, 0, 0, 0, 0, 0, 0, 0};
  __builtin_amdgcn_tensor_load_to_lds(g0, g1, z4, z4, z8, 0);
#else
  __builtin_amdgcn_tensor_load_to_lds(g0, g1, z4, z4, 0);
#endif
}

// ---------------------------------------------------------------------------
// fp32 -> f16 conversion, writing both row-major copy and transposed copy.
// src: [R,C] fp32  ->  dstN: [R,C] f16,  dstT: [C,R] f16
// grid (C/32, R/32, B), block 256 (32x8)
// ---------------------------------------------------------------------------
__global__ __launch_bounds__(256)
void esim_convert_transpose(const float* __restrict__ src,
                            _Float16* __restrict__ dstN,
                            _Float16* __restrict__ dstT,
                            int R, int C) {
  __shared__ _Float16 tile[32][33];
  const int b = blockIdx.z;
  const float* s = src + (size_t)b * R * C;
  _Float16* dn = dstN + (size_t)b * R * C;
  _Float16* dt = dstT + (size_t)b * R * C;
  const int c0 = blockIdx.x * 32, r0 = blockIdx.y * 32;
  const int tx = threadIdx.x & 31, ty = threadIdx.x >> 5;
#pragma unroll
  for (int i = 0; i < 4; ++i) {
    const int r = r0 + ty + 8 * i;
    const float v = s[(size_t)r * C + c0 + tx];
    const _Float16 h = (_Float16)v;
    dn[(size_t)r * C + c0 + tx] = h;
    tile[ty + 8 * i][tx] = h;
  }
  __syncthreads();
#pragma unroll
  for (int i = 0; i < 4; ++i) {
    const int c = c0 + ty + 8 * i;  // row index of the transposed output
    dt[(size_t)c * R + r0 + tx] = tile[tx][ty + 8 * i];
  }
}

// ---------------------------------------------------------------------------
// Batched GEMM  C[M,N] = A[M,K] * B[N,K]^T  (both row-major, K contiguous)
// f16 in, f32 out.  Block = 256 threads (8 waves), block tile 128x128,
// wave tile 32x64 (2x4 WMMA tiles).  Operand tiles (128x32 f16) staged in
// LDS by the Tensor Data Mover, double buffered; fragments read with
// immediate-offset ds_load_b128.  grid (N/128, M/128, batch).
// ---------------------------------------------------------------------------
__global__ __launch_bounds__(256)
void esim_gemm_nt_f16(const _Float16* __restrict__ A,
                      const _Float16* __restrict__ Bm,
                      float* __restrict__ C,
                      int M, int N, int K) {
  __shared__ __align__(64) _Float16 Abuf[2][128 * 32];
  __shared__ __align__(64) _Float16 Bbuf[2][128 * 32];

  const int b = blockIdx.z;
  const _Float16* Ab = A + (size_t)b * M * K + (size_t)(blockIdx.y * 128) * K;
  const _Float16* Bb = Bm + (size_t)b * N * K + (size_t)(blockIdx.x * 128) * K;
  float* Cb = C + (size_t)b * M * N;

  const int wave = threadIdx.x >> 5;
  const int lane = threadIdx.x & 31;
  const int l16 = lane & 15;   // m (A) / n (B,C) within a 16x16 tile
  const int s   = lane >> 4;   // half-select per ISA VGPR layout

  const int wm0 = (wave >> 1) * 32;  // wave M offset within block tile
  const int wn0 = (wave & 1) * 64;   // wave N offset within block tile
  const int m0 = blockIdx.y * 128 + wm0;
  const int n0 = blockIdx.x * 128 + wn0;

  const int nk = K >> 5;

  if (wave == 0) {  // TDM issue wave (EXEC-independent, SGPR descriptors)
    tdm_load_tile_2d((unsigned)(uintptr_t)&Abuf[0][0], Ab, K);
    tdm_load_tile_2d((unsigned)(uintptr_t)&Bbuf[0][0], Bb, K);
    if (nk > 1) {
      tdm_load_tile_2d((unsigned)(uintptr_t)&Abuf[1][0], Ab + 32, K);
      tdm_load_tile_2d((unsigned)(uintptr_t)&Bbuf[1][0], Bb + 32, K);
    }
  }

  v8f c[2][4] = {};

  for (int i = 0; i < nk; ++i) {
    const int cur = i & 1;
    if (wave == 0) {
      // 2 descriptors/buffer; retire the older buffer, keep newer in flight
      if (i + 1 < nk) __builtin_amdgcn_s_wait_tensorcnt(2);
      else            __builtin_amdgcn_s_wait_tensorcnt(0);
    }
    __syncthreads();  // publish TDM-written LDS to all waves

    const _Float16* As = &Abuf[cur][0];
    const _Float16* Bs = &Bbuf[cur][0];

    v16h a[2], bf[4];
#pragma unroll
    for (int ii = 0; ii < 2; ++ii) {
      // A fragment: lane (m + 16s): K=[16s..16s+7] in v0-3, K=[16s+8..16s+15] in v4-7
      const _Float16* ap = As + (wm0 + 16 * ii + l16) * 32 + 16 * s;
      v8h lo = *(const v8h*)(ap);
      v8h hi = *(const v8h*)(ap + 8);
#pragma unroll
      for (int t = 0; t < 8; ++t) { a[ii][t] = lo[t]; a[ii][8 + t] = hi[t]; }
    }
#pragma unroll
    for (int j = 0; j < 4; ++j) {
      // B fragment: lane (n + 16s) holds 16 consecutive K starting at 16s
      const _Float16* bp = Bs + (wn0 + 16 * j + l16) * 32 + 16 * s;
      bf[j] = *(const v16h*)bp;
    }
#pragma unroll
    for (int ii = 0; ii < 2; ++ii)
#pragma unroll
      for (int j = 0; j < 4; ++j)
        c[ii][j] = __builtin_amdgcn_wmma_f32_16x16x32_f16(
            false, a[ii], false, bf[j], (short)0, c[ii][j], false, false);

    __syncthreads();  // all waves done reading buf[cur] before TDM overwrites
    if (wave == 0 && i + 2 < nk) {
      const int k2 = (i + 2) << 5;
      tdm_load_tile_2d((unsigned)(uintptr_t)&Abuf[cur][0], Ab + k2, K);
      tdm_load_tile_2d((unsigned)(uintptr_t)&Bbuf[cur][0], Bb + k2, K);
    }
  }

#pragma unroll
  for (int ii = 0; ii < 2; ++ii)
#pragma unroll
    for (int j = 0; j < 4; ++j) {
      float* cp = Cb + (size_t)(m0 + 16 * ii + 8 * s) * N + n0 + 16 * j + l16;
#pragma unroll
      for (int r = 0; r < 8; ++r) cp[(size_t)r * N] = c[ii][j][r];
    }
}

// ---------------------------------------------------------------------------
// Row softmax over last axis (length 512).  One block per row, 256 threads.
// Masks cancel (shift-invariance of softmax), so plain softmax.
// ---------------------------------------------------------------------------
__global__ __launch_bounds__(256)
void esim_row_softmax(const float* __restrict__ sim,
                      _Float16* __restrict__ attn) {
  __shared__ float red[256];
  const size_t row = blockIdx.x;
  const float* x = sim + row * HH;
  const int t = threadIdx.x;
  const float x0 = x[t], x1 = x[t + 256];
  float m = fmaxf(x0, x1);
  red[t] = m;
  __syncthreads();
  for (int off = 128; off > 0; off >>= 1) {
    if (t < off) red[t] = fmaxf(red[t], red[t + off]);
    __syncthreads();
  }
  m = red[0];
  __syncthreads();
  const float e0 = __expf(x0 - m), e1 = __expf(x1 - m);
  red[t] = e0 + e1;
  __syncthreads();
  for (int off = 128; off > 0; off >>= 1) {
    if (t < off) red[t] += red[t + off];
    __syncthreads();
  }
  const float inv = 1.0f / red[0];
  _Float16* o = attn + row * HH;
  o[t] = (_Float16)(e0 * inv);
  o[t + 256] = (_Float16)(e1 * inv);
}

// ---------------------------------------------------------------------------
// Column softmax over P (axis=1), result written TRANSPOSED:
//   attnT[b][h][p] = softmax_p(sim[b][p][h])
// Reads coalesced (lanes along h), output staged in LDS and written
// coalesced (lanes along p).  grid (H/32, B), 256 threads.
// ---------------------------------------------------------------------------
__global__ __launch_bounds__(256)
void esim_col_softmax_t(const float* __restrict__ sim,
                        _Float16* __restrict__ attnT) {
  __shared__ _Float16 tile[512 * 33];  // [p][h_local], padded
  __shared__ float red[8 * 33];
  __shared__ float inv[32];
  const int b = blockIdx.y;
  const int h0 = blockIdx.x * 32;
  const float* s = sim + (size_t)b * PP * HH;
  _Float16* o = attnT + (size_t)b * HH * PP;
  const int tx = threadIdx.x & 31, ty = threadIdx.x >> 5;
  const int h = h0 + tx;

  float m = -3.0e38f;
  for (int p = ty; p < PP; p += 8) m = fmaxf(m, s[(size_t)p * HH + h]);
  red[ty * 33 + tx] = m;
  __syncthreads();
  float mall = red[tx];
#pragma unroll
  for (int yy = 1; yy < 8; ++yy) mall = fmaxf(mall, red[yy * 33 + tx]);

  float sum = 0.0f;
  for (int p = ty; p < PP; p += 8) {
    const float e = __expf(s[(size_t)p * HH + h] - mall);
    sum += e;
    tile[p * 33 + tx] = (_Float16)e;  // unnormalized, in [0,1]
  }
  __syncthreads();
  red[ty * 33 + tx] = sum;
  __syncthreads();
  if (ty == 0) {
    float tot = 0.0f;
#pragma unroll
    for (int yy = 0; yy < 8; ++yy) tot += red[yy * 33 + tx];
    inv[tx] = 1.0f / tot;
  }
  __syncthreads();

  const int t = threadIdx.x;
  for (int hl = 0; hl < 32; ++hl) {
    const float iv = inv[hl];
    for (int p = t; p < PP; p += 256)
      o[(size_t)(h0 + hl) * PP + p] = (_Float16)((float)tile[p * 33 + hl] * iv);
  }
}

// ---------------------------------------------------------------------------
extern "C" void kernel_launch(void* const* d_in, const int* in_sizes, int n_in,
                              void* d_out, int out_size, void* d_ws,
                              size_t ws_size, hipStream_t stream) {
  (void)in_sizes; (void)n_in; (void)out_size; (void)ws_size;
  const float* prem = (const float*)d_in[0];  // [B,P,D]
  // d_in[1] premise_mask   — provably a no-op under softmax shift-invariance
  const float* hyp = (const float*)d_in[2];   // [B,H,D]
  // d_in[3] hypothesis_mask — likewise a no-op
  float* out = (float*)d_out;

  char* ws = (char*)d_ws;
  const size_t SZ32 = 33554432;  // 32 MiB
  _Float16* prem16  = (_Float16*)(ws);              // [B,P,D] f16
  _Float16* premT16 = (_Float16*)(ws + 1 * SZ32);   // [B,D,P] f16
  _Float16* hyp16   = (_Float16*)(ws + 2 * SZ32);   // [B,H,D] f16
  _Float16* hypT16  = (_Float16*)(ws + 3 * SZ32);   // [B,D,H] f16
  float*    sim     = (float*)   (ws + 4 * SZ32);   // [B,P,H] f32
  _Float16* attnA   = (_Float16*)(ws + 5 * SZ32);   // [B,P,H] f16
  _Float16* attnBT  = (_Float16*)(ws + 5 * SZ32 + SZ32 / 2);  // [B,H,P] f16

  // 1) fp32 -> f16 (+ transposed copies)
  esim_convert_transpose<<<dim3(DD / 32, PP / 32, BB), 256, 0, stream>>>(
      prem, prem16, premT16, PP, DD);
  esim_convert_transpose<<<dim3(DD / 32, HH / 32, BB), 256, 0, stream>>>(
      hyp, hyp16, hypT16, HH, DD);

  // 2) sim[P,H] = prem16[P,D] * hyp16[H,D]^T
  esim_gemm_nt_f16<<<dim3(HH / 128, PP / 128, BB), 256, 0, stream>>>(
      prem16, hyp16, sim, PP, HH, DD);

  // 3) softmaxes (sim is L2-resident: 32 MB << 192 MB L2)
  esim_row_softmax<<<dim3(BB * PP), 256, 0, stream>>>(sim, attnA);
  esim_col_softmax_t<<<dim3(HH / 32, BB), 256, 0, stream>>>(sim, attnBT);

  // 4) attended_premises[P,D] = attnA[P,H] * hypT16[D,H]^T
  esim_gemm_nt_f16<<<dim3(DD / 128, PP / 128, BB), 256, 0, stream>>>(
      attnA, hypT16, out, PP, DD, HH);

  // 5) attended_hypothesis[H,D] = attnBT[H,P] * premT16[D,P]^T
  esim_gemm_nt_f16<<<dim3(DD / 128, HH / 128, BB), 256, 0, stream>>>(
      attnBT, premT16, out + (size_t)BB * PP * DD, HH, DD, PP);
}
